// NSRLoss_65781719105570
// MI455X (gfx1250) — matrix operand order repaired and checked by reading, not compiled, not run on hardware
//
#include <hip/hip_runtime.h>
#include <math.h>
#include <stdint.h>

typedef __attribute__((ext_vector_type(16))) _Float16 v16h;
typedef __attribute__((ext_vector_type(8)))  _Float16 v8h;
typedef __attribute__((ext_vector_type(8)))  float    v8f;
typedef __attribute__((ext_vector_type(4)))  unsigned int u32x4;
typedef __attribute__((ext_vector_type(8)))  int i32x8;
typedef __attribute__((ext_vector_type(4)))  int i32x4;

#define B_N   8192
#define L_N   2048
#define H_N   1024
#define NC    5
#define BETA_C 0.4f
#define EPS_C  0.05f

// LDS tile geometry: 64 rows x 32 halfs (64B) + 16B pad per row => 80B pitch (bank-conflict-free)
#define ROW_HALFS   40            // 80 bytes / 2
#define TILE_HALFS  (64 * ROW_HALFS)
#define TILE_BYTES  (TILE_HALFS * 2)   // 5120

// workspace byte offsets
static constexpr size_t XH_OFF  = 0;                                   // x   f16 [B][L]   32 MiB
static constexpr size_t W1T_OFF = XH_OFF  + (size_t)B_N * L_N * 2;     // W1T f16 [H][L]    4 MiB
static constexpr size_t W1R_OFF = W1T_OFF + (size_t)H_N * L_N * 2;     // W1  f16 [L][H]    4 MiB
static constexpr size_t H_OFF   = W1R_OFF + (size_t)L_N * H_N * 2;     // h   f32 [B][H]   32 MiB
static constexpr size_t ZY_OFF  = H_OFF   + (size_t)B_N * H_N * 4;     // z_y f32 [B]
static constexpr size_t WL1_OFF = ZY_OFF  + (size_t)B_N * 4;           // w_l1 f32 [B]
static constexpr size_t SUM_OFF = WL1_OFF + (size_t)B_N * 4;           // {mse, margin, correct}
static constexpr size_t VH_OFF  = SUM_OFF + 64;                        // V f16 [B][H]     16 MiB

union AFrag { v16h v; v8h h2[2]; };

__device__ __forceinline__ void lds_fence() { asm volatile("" ::: "memory"); }

// ---------------- TDM: 2D tile (64 rows x 32 halfs) global -> LDS, 16B pad per 64B row ----------------

__device__ __forceinline__ void tensor_load_tile(
    unsigned lds_byte_off, const void* gptr,
    unsigned row_stride_elems, unsigned tensor_d0, unsigned tensor_d1)
{
  unsigned long long ga = (unsigned long long)(uintptr_t)gptr;
  u32x4 g0 = { 1u,                                   // count=1, user descriptor
               lds_byte_off,                         // lds_addr
               (unsigned)(ga & 0xFFFFFFFFu),         // global_addr[31:0]
               (unsigned)(((ga >> 32) & 0x01FFFFFFu) | (2u << 30)) }; // addr[56:32] | type=2
  i32x8 g1 = { (int)((1u << 16)                      // data_size=1 (2 bytes)
                   | (1u << 20)                      // pad_enable
                   | (3u << 22)                      // pad_interval: 16 DWORDs (one 64B row)
                   | (3u << 25)),                    // pad_amount: 4 DWORDs (16B)
               (int)((tensor_d0 & 0xFFFFu) << 16),               // tensor_dim0 lo16
               (int)((tensor_d0 >> 16) | ((tensor_d1 & 0xFFFFu) << 16)),
               (int)((tensor_d1 >> 16) | (32u << 16)),           // tile_dim0 = 32 elems
               (int)64,                                          // tile_dim1 = 64 rows
               (int)row_stride_elems,                            // tensor_dim0_stride lo32
               0, 0 };
  i32x4 z4 = {0, 0, 0, 0};
#if __has_builtin(__builtin_amdgcn_tensor_load_to_lds)
 #if __has_include(<hip/amd_detail/amd_gfx1250_TDM.h>)
  i32x8 z8 = {0, 0, 0, 0, 0, 0, 0, 0};
  __builtin_amdgcn_tensor_load_to_lds(g0, g1, z4, z4, z8, 0);    // clang-23 / therock: 6 args
 #else
  __builtin_amdgcn_tensor_load_to_lds(g0, g1, z4, z4, 0);        // ROCm 7.2 clang-22: 5 args
 #endif
#endif
}

__device__ __forceinline__ void wait_tensorcnt0() {
#if __has_builtin(__builtin_amdgcn_s_wait_tensorcnt)
  __builtin_amdgcn_s_wait_tensorcnt((short)0);
#endif
}

// ---------------- shared TDM-pipelined WMMA core (two-phase ping-pong) ----------------
// btile[2][TILE_HALFS], A rows from arow (f16, stride = K), B tile rows from gB (f16, row stride bstride)

__device__ __forceinline__ void gemm_core(
    _Float16 (*btile)[TILE_HALFS], const _Float16* arow, const _Float16* gB,
    unsigned bstride, unsigned bdim1, int nstep,
    int tid, int sub, int l15, v8f acc[4])
{
  if (tid == 0) {
    tensor_load_tile(0u, gB, bstride, bstride, bdim1);
    wait_tensorcnt0();
  }
  __syncthreads();

  AFrag aA, aB;
  aA.h2[0] = *(const v8h*)(arow + 8 * sub);
  aA.h2[1] = *(const v8h*)(arow + 8 * sub + 16);

  for (int ks = 0; ks < nstep; ks += 2) {
    // ---- phase 0: compute buf0/aA, prefetch buf1/aB (always: nstep is even) ----
    if (tid == 0)
      tensor_load_tile((unsigned)TILE_BYTES, gB + (size_t)(ks + 1) * 32, bstride, bstride, bdim1);
    {
      int k0n = (ks + 1) * 32;
      aB.h2[0] = *(const v8h*)(arow + k0n + 8 * sub);
      aB.h2[1] = *(const v8h*)(arow + k0n + 8 * sub + 16);
    }
    lds_fence();
#pragma unroll
    for (int nt = 0; nt < 4; ++nt) {
      const _Float16* bp = &btile[0][(nt * 16 + l15) * ROW_HALFS + 8 * sub];
      AFrag bf;
      bf.h2[0] = *(const v8h*)(bp);
      bf.h2[1] = *(const v8h*)(bp + 16);
      acc[nt] = __builtin_amdgcn_wmma_f32_16x16x32_f16(
          false, aA.v, false, bf.v, (short)0, acc[nt], false, false);
    }
    if (tid == 0) wait_tensorcnt0();
    __syncthreads();

    // ---- phase 1: compute buf1/aB, prefetch buf0/aA ----
    bool more = (ks + 2 < nstep);
    if (more && tid == 0)
      tensor_load_tile(0u, gB + (size_t)(ks + 2) * 32, bstride, bstride, bdim1);
    if (more) {
      int k0n = (ks + 2) * 32;
      aA.h2[0] = *(const v8h*)(arow + k0n + 8 * sub);
      aA.h2[1] = *(const v8h*)(arow + k0n + 8 * sub + 16);
    }
    lds_fence();
#pragma unroll
    for (int nt = 0; nt < 4; ++nt) {
      const _Float16* bp = &btile[1][(nt * 16 + l15) * ROW_HALFS + 8 * sub];
      AFrag bf;
      bf.h2[0] = *(const v8h*)(bp);
      bf.h2[1] = *(const v8h*)(bp + 16);
      acc[nt] = __builtin_amdgcn_wmma_f32_16x16x32_f16(
          false, aB.v, false, bf.v, (short)0, acc[nt], false, false);
    }
    if (more) {
      if (tid == 0) wait_tensorcnt0();
      __syncthreads();
    }
  }
}

// ---------------- conversion / init ----------------

__global__ void convert_x_kernel(const float* __restrict__ x, _Float16* __restrict__ xh, int n) {
  int stride = gridDim.x * blockDim.x;
  for (int i = blockIdx.x * blockDim.x + threadIdx.x; i < n; i += stride)
    xh[i] = (_Float16)x[i];
}

__global__ void convert_w1_kernel(const float* __restrict__ W1,
                                  _Float16* __restrict__ w1r, _Float16* __restrict__ w1t) {
  int stride = gridDim.x * blockDim.x;
  for (int i = blockIdx.x * blockDim.x + threadIdx.x; i < L_N * H_N; i += stride) {
    float w = W1[i];
    int j = i / H_N;                            // input-dim row (0..2047)
    int c = i % H_N;                            // hidden col   (0..1023)
    w1r[i] = (_Float16)w;                       // [j][c], K-contiguous for GEMM2 B
    w1t[(size_t)c * L_N + j] = (_Float16)w;     // [c][j], K-contiguous for GEMM1 B
  }
}

__global__ void zero_kernel(float* __restrict__ wl1, float* __restrict__ sums) {
  int i = blockIdx.x * blockDim.x + threadIdx.x;
  if (i < B_N) wl1[i] = 0.0f;
  if (i < 4)   sums[i] = 0.0f;
}

// V[b,k] = (1 - h^2) * W2[k, y[b]]  (f16)
__global__ void build_v_kernel(const float* __restrict__ hbuf, const float* __restrict__ W2,
                               const int* __restrict__ y, _Float16* __restrict__ vbuf) {
  int stride = gridDim.x * blockDim.x;
  for (int i = blockIdx.x * blockDim.x + threadIdx.x; i < B_N * H_N; i += stride) {
    int b = i >> 10;            // / H_N
    int k = i & (H_N - 1);
    float hv = hbuf[i];
    vbuf[i] = (_Float16)((1.0f - hv * hv) * W2[k * NC + y[b]]);
  }
}

// ---------------- GEMM1: h = tanh(x @ W1 + b1) ----------------
// workgroup = 8 waves sharing one 64(N)x32(K) B tile staged by TDM (padded rows), double-buffered.

__global__ __launch_bounds__(256) void gemm1_tanh_kernel(
    const _Float16* __restrict__ xh, const _Float16* __restrict__ w1t,
    const float* __restrict__ b1, float* __restrict__ hbuf, int flag)
{
  __shared__ __align__(128) _Float16 btile[2][TILE_HALFS];

  int tid  = threadIdx.x;
  int wv   = tid >> 5;
  int lane = tid & 31;
  int sub  = lane >> 4;
  int l15  = lane & 15;

  int mg = blockIdx.x >> 4;               // 64 m-groups
  int ng = blockIdx.x & 15;               // 16 n-groups
  int m0 = (mg * 8 + wv) * 16;
  int n0 = ng * 64;
  int row = m0 + l15;

  // opaque (never-taken) store: makes the TDM-written LDS visible as modifiable to the compiler
  if (flag) { btile[0][tid] = (_Float16)1; btile[1][tid] = (_Float16)1; }

  v8f acc[4] = {};
  gemm_core(btile, xh + (size_t)row * L_N, w1t + (size_t)n0 * L_N,
            L_N, H_N, L_N / 32, tid, sub, l15, acc);

#pragma unroll
  for (int r = 0; r < 8; ++r) {
    int m = m0 + r + 8 * sub;
#pragma unroll
    for (int nt = 0; nt < 4; ++nt) {
      int n = n0 + nt * 16 + l15;
      float val = acc[nt][r] + b1[n];
      hbuf[(size_t)m * H_N + n] = tanhf(val);
    }
  }
}

// ---------------- GEMM2: G = V @ W1^T, w_l1[b] += sum_j |G[b,j]| ----------------

__global__ __launch_bounds__(256) void gemm2_absred_kernel(
    const _Float16* __restrict__ vbuf, const _Float16* __restrict__ w1r,
    float* __restrict__ wl1, int flag)
{
  __shared__ __align__(128) _Float16 btile[2][TILE_HALFS];

  int tid  = threadIdx.x;
  int wv   = tid >> 5;
  int lane = tid & 31;
  int sub  = lane >> 4;
  int l15  = lane & 15;

  int mg = blockIdx.x >> 5;               // 64 m-groups
  int ng = blockIdx.x & 31;               // 32 n-groups
  int m0 = (mg * 8 + wv) * 16;
  int n0 = ng * 64;
  int row = m0 + l15;

  if (flag) { btile[0][tid] = (_Float16)1; btile[1][tid] = (_Float16)1; }

  v8f acc[4] = {};
  gemm_core(btile, vbuf + (size_t)row * H_N, w1r + (size_t)n0 * H_N,
            H_N, L_N, H_N / 32, tid, sub, l15, acc);

#pragma unroll
  for (int r = 0; r < 8; ++r) {
    float s = fabsf(acc[0][r]) + fabsf(acc[1][r]) + fabsf(acc[2][r]) + fabsf(acc[3][r]);
    s += __shfl_xor(s, 1, 32);
    s += __shfl_xor(s, 2, 32);
    s += __shfl_xor(s, 4, 32);
    s += __shfl_xor(s, 8, 32);
    if (l15 == 0) atomicAdd(&wl1[m0 + r + 8 * sub], s);
  }
}

// ---------------- head: logits, mse, margin, correct, z_y ----------------

__global__ __launch_bounds__(256) void head_kernel(
    const float* __restrict__ hbuf, const float* __restrict__ W2,
    const float* __restrict__ b2, const int* __restrict__ y,
    float* __restrict__ zybuf, float* __restrict__ sums)
{
  int b    = (blockIdx.x * blockDim.x + threadIdx.x) >> 5;
  int lane = threadIdx.x & 31;
  const float* hrow = hbuf + (size_t)b * H_N;

  float c[NC] = {0.f, 0.f, 0.f, 0.f, 0.f};
  for (int k = lane; k < H_N; k += 32) {
    float hv = hrow[k];
#pragma unroll
    for (int j = 0; j < NC; ++j) c[j] += hv * W2[k * NC + j];
  }
#pragma unroll
  for (int j = 0; j < NC; ++j) {
    c[j] += __shfl_xor(c[j], 1, 32);
    c[j] += __shfl_xor(c[j], 2, 32);
    c[j] += __shfl_xor(c[j], 4, 32);
    c[j] += __shfl_xor(c[j], 8, 32);
    c[j] += __shfl_xor(c[j], 16, 32);
  }
  if (lane == 0) {
    int yi = y[b];
    float logit[NC];
#pragma unroll
    for (int j = 0; j < NC; ++j) logit[j] = c[j] + b2[j];
    float zy = logit[yi];
    float mse = 0.f, margin = 0.f, mx = logit[0];
    int amax = 0;
#pragma unroll
    for (int j = 0; j < NC; ++j) {
      float t = logit[j] - ((j == yi) ? 1.0f : 0.0f);
      mse += t * t;
      if (j != yi) margin += fmaxf(0.0f, 1.0f - zy + logit[j]);
      if (logit[j] > mx) { mx = logit[j]; amax = j; }
    }
    zybuf[b] = zy;
    atomicAdd(&sums[0], mse);
    atomicAdd(&sums[1], margin);
    atomicAdd(&sums[2], (amax == yi) ? 1.0f : 0.0f);
  }
}

// ---------------- final combine ----------------

__global__ __launch_bounds__(256) void final_kernel(
    const float* __restrict__ wl1, const float* __restrict__ zybuf,
    const float* __restrict__ sums, float* __restrict__ out)
{
  __shared__ float red[256];
  float local = 0.0f;
  for (int b = threadIdx.x; b < B_N; b += 256) {
    float R = wl1[b] * EPS_C / (fabsf(zybuf[b]) + 1e-8f);
    local += log1pf(R);
  }
  red[threadIdx.x] = local;
  __syncthreads();
  for (int s = 128; s > 0; s >>= 1) {
    if (threadIdx.x < s) red[threadIdx.x] += red[threadIdx.x + s];
    __syncthreads();
  }
  if (threadIdx.x == 0) {
    float nsr    = BETA_C * red[0] / (float)B_N;
    float mse    = sums[0] / (float)(B_N * NC);
    float margin = sums[1] / (float)B_N;
    float cmask  = sums[2] / (float)B_N;
    out[0] = mse + (margin + nsr) * cmask;
  }
}

// ---------------- launch ----------------

extern "C" void kernel_launch(void* const* d_in, const int* in_sizes, int n_in,
                              void* d_out, int out_size, void* d_ws, size_t ws_size,
                              hipStream_t stream) {
  const float* x  = (const float*)d_in[0];
  const int*   y  = (const int*)d_in[1];
  const float* W1 = (const float*)d_in[2];
  const float* b1 = (const float*)d_in[3];
  const float* W2 = (const float*)d_in[4];
  const float* b2 = (const float*)d_in[5];

  char* ws = (char*)d_ws;
  _Float16* xh   = (_Float16*)(ws + XH_OFF);
  _Float16* w1t  = (_Float16*)(ws + W1T_OFF);
  _Float16* w1r  = (_Float16*)(ws + W1R_OFF);
  float*    hbuf = (float*)(ws + H_OFF);
  float*    zyb  = (float*)(ws + ZY_OFF);
  float*    wl1  = (float*)(ws + WL1_OFF);
  float*    sums = (float*)(ws + SUM_OFF);
  _Float16* vbuf = (_Float16*)(ws + VH_OFF);
  float*    out  = (float*)d_out;

  convert_x_kernel<<<1024, 256, 0, stream>>>(x, xh, B_N * L_N);
  convert_w1_kernel<<<1024, 256, 0, stream>>>(W1, w1r, w1t);
  zero_kernel<<<(B_N + 255) / 256, 256, 0, stream>>>(wl1, sums);

  // GEMM1: 64 m-groups * 16 n-groups, 8 waves each
  gemm1_tanh_kernel<<<64 * 16, 256, 0, stream>>>(xh, w1t, b1, hbuf, 0);
  // head: one wave per sample
  head_kernel<<<B_N * 32 / 256, 256, 0, stream>>>(hbuf, W2, b2, y, zyb, sums);
  // V = (1-h^2) * W2[:,y]
  build_v_kernel<<<4096, 256, 0, stream>>>(hbuf, W2, y, vbuf);
  // GEMM2: 64 m-groups * 32 n-groups
  gemm2_absred_kernel<<<64 * 32, 256, 0, stream>>>(vbuf, w1r, wl1, 0);

  final_kernel<<<1, 256, 0, stream>>>(wl1, zyb, sums, out);
}
